// GAT_width_69277822484764
// MI455X (gfx1250) — matrix operand order repaired and checked by reading, not compiled, not run on hardware
//
#include <hip/hip_runtime.h>

// ---------------------------------------------------------------------------
// 2-layer GAT for MI455X (gfx1250): bf16 WMMA GEMMs + atomic edge softmax/agg
// ---------------------------------------------------------------------------

typedef __attribute__((ext_vector_type(16))) __bf16 v16bf;
typedef __attribute__((ext_vector_type(8)))  float  v8f;

#define N_NODES   20000
#define E_EDGES   320000
#define EP_EDGES  340000        // E + N self loops
#define F_IN      256
#define NHEAD1    8
#define NHID      64
#define HC1       512           // NHEAD1*NHID
#define C2        40
#define C2P       48            // padded to 3 n-tiles of 16
#define NEG_SLOPE 0.2f

#define MT   1250               // 20000 / 16
#define KT1  8                  // 256 / 32
#define NT1  32                 // 512 / 16
#define KT2  16                 // 512 / 32
#define NT2  3                  // 48 / 16

#define ENC_NEG_INF 0x007FFFFFu // order-preserving encoding of -inf

__device__ __forceinline__ __bf16 f2bf(float f) {
  unsigned u = __float_as_uint(f);
  unsigned r = (u + 0x7fffu + ((u >> 16) & 1u)) >> 16;  // RNE
  unsigned short h = (unsigned short)r;
  return __builtin_bit_cast(__bf16, h);
}
// monotonic float <-> uint mapping for atomicMax on signed floats
__device__ __forceinline__ unsigned enc_ord(float x) {
  unsigned u = __float_as_uint(x);
  return (u & 0x80000000u) ? ~u : (u | 0x80000000u);
}
__device__ __forceinline__ float dec_ord(unsigned u) {
  return (u & 0x80000000u) ? __uint_as_float(u & 0x7fffffffu)
                           : __uint_as_float(~u);
}

// -------------------------------- utility ----------------------------------
__global__ void k_fill_u32(unsigned* __restrict__ p, long long n, unsigned v) {
  long long i = (long long)blockIdx.x * blockDim.x + threadIdx.x;
  long long stride = (long long)gridDim.x * blockDim.x;
  for (; i < n; i += stride) p[i] = v;
}

// ---------------------- fragment repack (A: 16x32 bf16) --------------------
// dst[(mt*ktiles+kt)*32 + lane][16] : lane m=lane&15, half=lane>>4
//   elem j in 0..7  -> K = kt*32 + half*16 + j
//   elem j in 8..15 -> K = kt*32 + half*16 + j
__global__ void k_pack_a_f32(const float* __restrict__ X, __bf16* __restrict__ dst,
                             int mtiles, int ktiles, int ldk) {
  int t = blockIdx.x * blockDim.x + threadIdx.x;
  int total = mtiles * ktiles * 32;
  if (t >= total) return;
  int lane = t & 31;
  int kt   = (t >> 5) % ktiles;
  int mt   = (t >> 5) / ktiles;
  int m    = mt * 16 + (lane & 15);
  int half = lane >> 4;
  const float* row = X + (long long)m * ldk + kt * 32;
  v16bf o;
#pragma unroll
  for (int j = 0; j < 8; ++j) o[j] = f2bf(row[half * 16 + j]);
#pragma unroll
  for (int j = 0; j < 8; ++j) o[8 + j] = f2bf(row[half * 16 + 8 + j]);
  *(v16bf*)(dst + (long long)t * 16) = o;
}

// ---------------------- fragment repack (B: 32x16 bf16) --------------------
// dst[(kt*ntiles+nt)*32 + lane][16] : col = nt*16 + (lane&15)
//   elem j -> K = kt*32 + (lane>>4)*16 + j ;  zero-pad cols >= ncols
__global__ void k_pack_b_f32(const float* __restrict__ W, __bf16* __restrict__ dst,
                             int ktiles, int ntiles, int ncols) {
  int t = blockIdx.x * blockDim.x + threadIdx.x;
  int total = ktiles * ntiles * 32;
  if (t >= total) return;
  int lane = t & 31;
  int nt   = (t >> 5) % ntiles;
  int kt   = (t >> 5) / ntiles;
  int col  = nt * 16 + (lane & 15);
  int kbase = kt * 32 + (lane >> 4) * 16;
  v16bf o;
#pragma unroll
  for (int j = 0; j < 16; ++j) {
    float v = (col < ncols) ? W[(long long)(kbase + j) * ncols + col] : 0.f;
    o[j] = f2bf(v);
  }
  *(v16bf*)(dst + (long long)t * 16) = o;
}

// ------------------------------ WMMA GEMM ----------------------------------
// one wave32 per 16x16 output tile; KTILES chained v_wmma_f32_16x16x32_bf16
template <int KTILES>
__global__ void k_wmma_gemm(const __bf16* __restrict__ Apack,
                            const __bf16* __restrict__ Bpack,
                            float* __restrict__ C,
                            int ntiles, int ldc, int totWaves) {
  int wid  = (int)((blockIdx.x * blockDim.x + threadIdx.x) >> 5);
  int lane = threadIdx.x & 31;
  if (wid >= totWaves) return;
  int mt = wid / ntiles;
  int nt = wid % ntiles;
  const __bf16* aptr = Apack + ((long long)mt * KTILES * 32 + lane) * 16;
  const __bf16* bptr = Bpack + ((long long)nt * 32 + lane) * 16;
  const long long bstride = (long long)ntiles * 512;  // 32 lanes * 16 elems
  v8f acc = {0.f, 0.f, 0.f, 0.f, 0.f, 0.f, 0.f, 0.f};
#pragma unroll
  for (int kt = 0; kt < KTILES; ++kt) {
    v16bf a = *(const v16bf*)aptr;
    v16bf b = *(const v16bf*)bptr;
    acc = __builtin_amdgcn_wmma_f32_16x16x32_bf16(false, a, false, b,
                                                  (short)0, acc, false, false);
    aptr += 512;
    bptr += bstride;
  }
  // C/D layout: VGPR v, lane l -> row = v + (l>>4)*8, col = l&15
  int col  = nt * 16 + (lane & 15);
  int row0 = mt * 16 + (lane >> 4) * 8;
  float* cp = C + (long long)row0 * ldc + col;
#pragma unroll
  for (int v = 0; v < 8; ++v) cp[(long long)v * ldc] = acc[v];
}

// ----------------------- attention coefficient dot -------------------------
__global__ void k_alpha(const float* __restrict__ H, const float* __restrict__ Asrc,
                        const float* __restrict__ Adst, float* __restrict__ os,
                        float* __restrict__ od, int nh, int cdim, int ld, int total) {
  int t = blockIdx.x * blockDim.x + threadIdx.x;
  if (t >= total) return;
  int n = t / nh, h = t % nh;
  const float* row = H + (long long)n * ld + h * cdim;
  const float* vs = Asrc + h * cdim;
  const float* vd = Adst + h * cdim;
  float s = 0.f, d = 0.f;
  for (int c = 0; c < cdim; ++c) {
    float x = row[c];
    s = fmaf(x, vs[c], s);
    d = fmaf(x, vd[c], d);
  }
  os[t] = s;
  od[t] = d;
}

// ------------------------------ edge passes --------------------------------
template <int NH>
__global__ void k_edge_max(const int* __restrict__ ei, const float* __restrict__ as,
                           const float* __restrict__ ad, unsigned* __restrict__ emax) {
  int t = blockIdx.x * blockDim.x + threadIdx.x;
  if (t >= EP_EDGES * NH) return;
  int e = t / NH, h = t % NH;
  int s = (e < E_EDGES) ? ei[e] : (e - E_EDGES);
  int d = (e < E_EDGES) ? ei[E_EDGES + e] : (e - E_EDGES);
  float x = as[s * NH + h] + ad[d * NH + h];
  x = (x > 0.f) ? x : NEG_SLOPE * x;
  atomicMax(&emax[d * NH + h], enc_ord(x));
}

template <int NH>
__global__ void k_edge_expsum(const int* __restrict__ ei, const float* __restrict__ as,
                              const float* __restrict__ ad,
                              const unsigned* __restrict__ emax,
                              float* __restrict__ aE, float* __restrict__ den) {
  int t = blockIdx.x * blockDim.x + threadIdx.x;
  if (t >= EP_EDGES * NH) return;
  int e = t / NH, h = t % NH;
  int s = (e < E_EDGES) ? ei[e] : (e - E_EDGES);
  int d = (e < E_EDGES) ? ei[E_EDGES + e] : (e - E_EDGES);
  float x = as[s * NH + h] + ad[d * NH + h];
  x = (x > 0.f) ? x : NEG_SLOPE * x;
  float ee = __expf(x - dec_ord(emax[d * NH + h]));
  aE[t] = ee;
  atomicAdd(&den[d * NH + h], ee);
}

template <int NH>
__global__ void k_edge_norm(const int* __restrict__ ei, const float* __restrict__ den,
                            float* __restrict__ aE) {
  int t = blockIdx.x * blockDim.x + threadIdx.x;
  if (t >= EP_EDGES * NH) return;
  int e = t / NH, h = t % NH;
  int d = (e < E_EDGES) ? ei[E_EDGES + e] : (e - E_EDGES);
  aE[t] = aE[t] / (den[d * NH + h] + 1e-16f);
}

// one thread per (edge, channel); CTOT = 1<<LG2C padded channel count
template <int NH, int CPH, int LG2C, int LDH, int LDO>
__global__ void k_edge_agg(const int* __restrict__ ei, const float* __restrict__ aE,
                           const float* __restrict__ H, float* __restrict__ O) {
  long long t = (long long)blockIdx.x * blockDim.x + threadIdx.x;
  if (t >= ((long long)EP_EDGES << LG2C)) return;
  int c = (int)(t & ((1 << LG2C) - 1));
  if (c >= NH * CPH) return;
  int e = (int)(t >> LG2C);
  int h = c / CPH;
  int s = (e < E_EDGES) ? ei[e] : (e - E_EDGES);
  int d = (e < E_EDGES) ? ei[E_EDGES + e] : (e - E_EDGES);
  float a = aE[(long long)e * NH + h];
  atomicAdd(&O[(long long)d * LDO + c], a * H[(long long)s * LDH + c]);
}

// ------------------------------ epilogues ----------------------------------
__global__ void k_bias_elu(float* __restrict__ o, const float* __restrict__ b) {
  long long t = (long long)blockIdx.x * blockDim.x + threadIdx.x;
  if (t >= (long long)N_NODES * HC1) return;
  int c = (int)(t & (HC1 - 1));
  float x = o[t] + b[c];
  o[t] = (x > 0.f) ? x : (__expf(x) - 1.f);
}

__global__ void k_logsoftmax(const float* __restrict__ o2, const float* __restrict__ b2,
                             float* __restrict__ out) {
  int n = blockIdx.x * blockDim.x + threadIdx.x;
  if (n >= N_NODES) return;
  float l[C2];
  float m = -3.4e38f;
#pragma unroll
  for (int c = 0; c < C2; ++c) {
    l[c] = o2[(long long)n * C2 + c] + b2[c];
    m = fmaxf(m, l[c]);
  }
  float s = 0.f;
#pragma unroll
  for (int c = 0; c < C2; ++c) s += __expf(l[c] - m);
  float lse = m + __logf(s);
#pragma unroll
  for (int c = 0; c < C2; ++c) out[(long long)n * C2 + c] = l[c] - lse;
}

// ------------------------------- launcher ----------------------------------
extern "C" void kernel_launch(void* const* d_in, const int* in_sizes, int n_in,
                              void* d_out, int out_size, void* d_ws, size_t ws_size,
                              hipStream_t stream) {
  (void)in_sizes; (void)n_in; (void)out_size; (void)ws_size;
  const float* x      = (const float*)d_in[0];
  const int*   ei     = (const int*)d_in[1];   // [2][E] src row then dst row
  const float* W1     = (const float*)d_in[2];
  const float* att_s1 = (const float*)d_in[3];
  const float* att_d1 = (const float*)d_in[4];
  const float* b1     = (const float*)d_in[5];
  const float* W2     = (const float*)d_in[6];
  const float* att_s2 = (const float*)d_in[7];
  const float* att_d2 = (const float*)d_in[8];
  const float* b2     = (const float*)d_in[9];
  float* out = (float*)d_out;

  char* ws = (char*)d_ws;
  size_t off = 0;
  auto take = [&](size_t bytes) -> void* {
    void* p = ws + off;
    off += (bytes + 255) & ~(size_t)255;
    return p;
  };
  __bf16*   xpack  = (__bf16*)  take((size_t)MT * KT1 * 32 * 16 * 2);     // 10.24 MB
  __bf16*   w1pack = (__bf16*)  take((size_t)KT1 * NT1 * 32 * 16 * 2);    // 256 KB
  __bf16*   w2pack = (__bf16*)  take((size_t)KT2 * NT2 * 32 * 16 * 2);    // 48 KB
  float*    h1     = (float*)   take((size_t)N_NODES * HC1 * 4);          // 40.96 MB
  float*    asv1   = (float*)   take((size_t)N_NODES * NHEAD1 * 4);
  float*    adv1   = (float*)   take((size_t)N_NODES * NHEAD1 * 4);
  unsigned* em1    = (unsigned*)take((size_t)N_NODES * NHEAD1 * 4);
  float*    den1   = (float*)   take((size_t)N_NODES * NHEAD1 * 4);
  float*    aE1    = (float*)   take((size_t)EP_EDGES * NHEAD1 * 4);      // 10.88 MB
  float*    out1   = (float*)   take((size_t)N_NODES * HC1 * 4);          // 40.96 MB
  float*    asv2   = (float*)   take((size_t)N_NODES * 4);
  float*    adv2   = (float*)   take((size_t)N_NODES * 4);
  unsigned* em2    = (unsigned*)take((size_t)N_NODES * 4);
  float*    den2   = (float*)   take((size_t)N_NODES * 4);
  float*    aE2    = (float*)   take((size_t)EP_EDGES * 4);
  // lifetime-safe aliases (stream-ordered):
  __bf16* h2pack = (__bf16*)h1;   // h1 dead after layer-1 aggregation
  float*  h2     = aE1;           // aE1 dead after layer-1 aggregation (960k <= 2.72M f32)
  float*  out2   = (float*)xpack; // xpack dead after GEMM1 (800k f32 <= 10.24 MB)

  const int B = 256;
  auto cdiv = [](long long a, long long b) { return (int)((a + b - 1) / b); };

  // ---- init (note: out2 aliases xpack -> its zero-fill happens AFTER GEMM1)
  k_fill_u32<<<1024, B, 0, stream>>>(em1, (long long)N_NODES * NHEAD1, ENC_NEG_INF);
  k_fill_u32<<<1024, B, 0, stream>>>((unsigned*)den1, (long long)N_NODES * NHEAD1, 0u);
  k_fill_u32<<<4096, B, 0, stream>>>((unsigned*)out1, (long long)N_NODES * HC1, 0u);
  k_fill_u32<<<512,  B, 0, stream>>>(em2, N_NODES, ENC_NEG_INF);
  k_fill_u32<<<512,  B, 0, stream>>>((unsigned*)den2, N_NODES, 0u);

  // ---- layer 1: h1 = x @ W1 (bf16 WMMA, f32 accumulate)
  k_pack_a_f32<<<cdiv((long long)MT * KT1 * 32, B), B, 0, stream>>>(x, xpack, MT, KT1, F_IN);
  k_pack_b_f32<<<cdiv((long long)KT1 * NT1 * 32, B), B, 0, stream>>>(W1, w1pack, KT1, NT1, HC1);
  k_pack_b_f32<<<cdiv((long long)KT2 * NT2 * 32, B), B, 0, stream>>>(W2, w2pack, KT2, NT2, C2);
  k_wmma_gemm<KT1><<<cdiv((long long)MT * NT1 * 32, B), B, 0, stream>>>(
      xpack, w1pack, h1, NT1, HC1, MT * NT1);

  // xpack now dead -> safe to zero its alias out2
  k_fill_u32<<<1024, B, 0, stream>>>((unsigned*)out2, (long long)N_NODES * C2, 0u);

  // ---- layer 1 attention + aggregation
  k_alpha<<<cdiv((long long)N_NODES * NHEAD1, B), B, 0, stream>>>(
      h1, att_s1, att_d1, asv1, adv1, NHEAD1, NHID, HC1, N_NODES * NHEAD1);
  k_edge_max<NHEAD1><<<cdiv((long long)EP_EDGES * NHEAD1, B), B, 0, stream>>>(ei, asv1, adv1, em1);
  k_edge_expsum<NHEAD1><<<cdiv((long long)EP_EDGES * NHEAD1, B), B, 0, stream>>>(
      ei, asv1, adv1, em1, aE1, den1);
  k_edge_norm<NHEAD1><<<cdiv((long long)EP_EDGES * NHEAD1, B), B, 0, stream>>>(ei, den1, aE1);
  k_edge_agg<NHEAD1, NHID, 9, HC1, HC1><<<cdiv((long long)EP_EDGES * 512, B), B, 0, stream>>>(
      ei, aE1, h1, out1);
  k_bias_elu<<<cdiv((long long)N_NODES * HC1, B), B, 0, stream>>>(out1, b1);

  // ---- layer 2: h2 = elu_out @ W2 (padded 40->48 cols)
  k_pack_a_f32<<<cdiv((long long)MT * KT2 * 32, B), B, 0, stream>>>(out1, h2pack, MT, KT2, HC1);
  k_wmma_gemm<KT2><<<cdiv((long long)MT * NT2 * 32, B), B, 0, stream>>>(
      h2pack, w2pack, h2, NT2, C2P, MT * NT2);

  // ---- layer 2 attention + aggregation (H=1)
  k_alpha<<<cdiv(N_NODES, B), B, 0, stream>>>(h2, att_s2, att_d2, asv2, adv2, 1, C2, C2P, N_NODES);
  k_edge_max<1><<<cdiv(EP_EDGES, B), B, 0, stream>>>(ei, asv2, adv2, em2);
  k_edge_expsum<1><<<cdiv(EP_EDGES, B), B, 0, stream>>>(ei, asv2, adv2, em2, aE2, den2);
  k_edge_norm<1><<<cdiv(EP_EDGES, B), B, 0, stream>>>(ei, den2, aE2);
  k_edge_agg<1, C2, 6, C2P, C2><<<cdiv((long long)EP_EDGES * 64, B), B, 0, stream>>>(
      ei, aE2, h2, out2);

  // ---- bias + log_softmax
  k_logsoftmax<<<cdiv(N_NODES, B), B, 0, stream>>>(out2, b2, out);
}
